// Net_21663815041319
// MI455X (gfx1250) — matrix operand order repaired and checked by reading, not compiled, not run on hardware
//
#include <hip/hip_runtime.h>
#include <hip/hip_bf16.h>
#include <math.h>

typedef __attribute__((ext_vector_type(16))) _Float16 v16h;
typedef __attribute__((ext_vector_type(8)))  float    v8f;

#define NN   65536      // B*NPG
#define EE   2097152    // total edges
#define BB   64
#define NPGC 1024
#define EPG  32768      // edges per graph

// ---------------- workspace layout (float offsets) ----------------
static const size_t OFF_AGG0 = 0;                                  // N*64
static const size_t OFF_X1C  = OFF_AGG0 + (size_t)NN * 64;         // N*96 concat x11|x12|x13, cols 90..95 zero
static const size_t OFF_AGGH = OFF_X1C  + (size_t)NN * 96;         // N*32 scratch agg
static const size_t OFF_SBP  = OFF_AGGH + (size_t)NN * 32;         // N*16 (9 used: q2|q3|q4)
static const size_t OFF_FLD  = OFF_SBP  + (size_t)NN * 16;         // N*16
static const size_t OFF_SC   = OFF_FLD  + (size_t)NN * 16;         // N*16 cluster softmax
static const size_t OFF_DOUT = OFF_SC   + (size_t)NN * 16;         // N
static const size_t OFF_DIN  = OFF_DOUT + (size_t)NN;              // N
static const size_t OFF_P1X  = OFF_DIN  + (size_t)NN;              // 64*16*96 (atomic-accumulated)
static const size_t OFF_PADJ = OFF_P1X  + (size_t)64 * 16 * 96;    // 64*100
static const size_t OFF_X2O  = OFF_PADJ + (size_t)64 * 100;        // 64*96
static const size_t OFF_X1O  = OFF_X2O  + (size_t)64 * 96;         // 64*96
static const size_t OFF_SCAL = OFF_X1O  + (size_t)64 * 96;         // 32 scalars
static const size_t OFF_X1P  = OFF_SCAL + 32;                      // 64*8*96 partial maxes

// ---------------- utility ----------------
__global__ void zero_kernel(float* __restrict__ p, size_t n) {
  size_t i = (size_t)blockIdx.x * blockDim.x + threadIdx.x;
  if (i < n) p[i] = 0.0f;
}

// ---------------- GIN neighbor sum: agg[dst] += x[src], compile-time shapes ----------------
template <int SX, int NC>
__global__ void scatter_add_kernel(const int* __restrict__ src, const int* __restrict__ dst,
                                   const float* __restrict__ xin, float* __restrict__ agg) {
  int e = blockIdx.x * blockDim.x + threadIdx.x;
  if (e >= EE) return;
  int s = src[e], d = dst[e];
  const float* xr = xin + (size_t)s * SX;  // xin may be pre-offset by an even column base
  float* ar = agg + (size_t)d * NC;
  #pragma unroll
  for (int p = 0; p < NC / 2; ++p) {
    float2 v = *(const float2*)(xr + 2 * p);
    atomicAdd(ar + 2 * p + 0, v.x);
    atomicAdd(ar + 2 * p + 1, v.y);
  }
}

// ---------------- GIN MLP: out = relu((x+agg)@W1) @ W2, WMMA f16 ----------------
// One wave = 16 nodes, 4 waves/block. All bounds compile-time: branch-free fragment builds.
// Writes result into x1cat at column CO (stride 96), zero-spilling pad columns.
template <int SX, int SA, int KVALID, int CO>
__global__ void gin_mlp_kernel(const float* __restrict__ xin,  // row stride SX (pre-offset base col)
                               const float* __restrict__ agg,  // row stride SA
                               const float* __restrict__ W1,   // [KVALID x 30]
                               const float* __restrict__ W2,   // [30 x 30]
                               float* __restrict__ out) {      // x1cat, stride 96
  __shared__ _Float16 lds[4][16 * 32];
  constexpr int KSTEPS = (KVALID + 31) / 32;
  int wave = threadIdx.x >> 5;
  int lane = threadIdx.x & 31;
  int tile = blockIdx.x * 4 + wave;
  int row = lane & 15, hi = lane >> 4, col = lane & 15;
  int node = tile * 16 + row;
  const float* xr = xin + (size_t)node * SX;
  const float* ar = agg + (size_t)node * SA;
  v8f c0 = {}; v8f c1 = {};
  #pragma unroll
  for (int ks = 0; ks < KSTEPS; ++ks) {
    int kb = ks * 32 + hi * 8;
    v16h a;
    #pragma unroll
    for (int p = 0; p < 4; ++p) {  // elements 0..7 : K = kb + 0..7
      float2 xv = *(const float2*)(xr + kb + 2 * p);
      float2 av = *(const float2*)(ar + kb + 2 * p);
      a[2 * p + 0] = (_Float16)(xv.x + av.x);
      a[2 * p + 1] = (_Float16)(xv.y + av.y);
    }
    #pragma unroll
    for (int p = 0; p < 4; ++p) {  // elements 8..15 : K = kb + 16..23
      float2 xv = *(const float2*)(xr + kb + 16 + 2 * p);
      float2 av = *(const float2*)(ar + kb + 16 + 2 * p);
      a[8 + 2 * p + 0] = (_Float16)(xv.x + av.x);
      a[8 + 2 * p + 1] = (_Float16)(xv.y + av.y);
    }
    v16h b0, b1;
    int c1i = (col + 16 < 30) ? col + 16 : 29;  // clamp, select below
    #pragma unroll
    for (int e = 0; e < 16; ++e) {
      int k = ks * 32 + hi * 8 + (e < 8 ? e : e + 8);
      int kc = (k < KVALID) ? k : (KVALID - 1);
      float w0 = W1[kc * 30 + col];
      float w1 = W1[kc * 30 + c1i];
      b0[e] = (_Float16)((k < KVALID) ? w0 : 0.0f);
      b1[e] = (_Float16)((k < KVALID && col + 16 < 30) ? w1 : 0.0f);
    }
    c0 = __builtin_amdgcn_wmma_f32_16x16x32_f16(false, a, false, b0, (short)0, c0, false, false);
    c1 = __builtin_amdgcn_wmma_f32_16x16x32_f16(false, a, false, b1, (short)0, c1, false, false);
  }
  _Float16* L = lds[wave];
  #pragma unroll
  for (int r = 0; r < 8; ++r) {
    int orow = r + 8 * hi;
    L[orow * 32 + col]      = (_Float16)fmaxf(c0[r], 0.0f);
    L[orow * 32 + 16 + col] = (_Float16)fmaxf(c1[r], 0.0f);  // cols>=30 exact zeros
  }
  __syncthreads();
  v16h a2;
  #pragma unroll
  for (int e = 0; e < 16; ++e) {
    int k = hi * 8 + (e < 8 ? e : e + 8);
    a2[e] = L[row * 32 + k];
  }
  v16h b20, b21;
  {
    int c1i = (col + 16 < 30) ? col + 16 : 29;
    #pragma unroll
    for (int e = 0; e < 16; ++e) {
      int k = hi * 8 + (e < 8 ? e : e + 8);
      int kc = (k < 30) ? k : 29;
      float w0 = W2[kc * 30 + col];
      float w1 = W2[kc * 30 + c1i];
      b20[e] = (_Float16)((k < 30) ? w0 : 0.0f);
      b21[e] = (_Float16)((k < 30 && col + 16 < 30) ? w1 : 0.0f);
    }
  }
  v8f z = {};
  v8f d0 = __builtin_amdgcn_wmma_f32_16x16x32_f16(false, a2, false, b20, (short)0, z, false, false);
  v8f d1 = __builtin_amdgcn_wmma_f32_16x16x32_f16(false, a2, false, b21, (short)0, z, false, false);
  #pragma unroll
  for (int r = 0; r < 8; ++r) {
    int onode = tile * 16 + r + 8 * hi;
    float* orow = out + (size_t)onode * 96 + CO;
    orow[col]      = d0[r];
    orow[16 + col] = (col + 16 < 30) ? d1[r] : 0.0f;  // zero-spill pad cols
    if constexpr (CO == 60) {
      if (col < 4) orow[32 + col] = 0.0f;             // zero cols 92..95
    }
  }
}

// ---------------- belief propagation ----------------
__global__ void bp_init_kernel(float* __restrict__ s) {
  int n = blockIdx.x * blockDim.x + threadIdx.x;
  if (n >= NN) return;
  float* o = s + (size_t)n * 16;
  float nf = (float)(n + 1);
  int base = 0;
  for (int q = 2; q <= 4; ++q) {
    float l[4]; float mx = -1e30f;
    for (int k = 0; k < q; ++k) { l[k] = 0.01f * __sinf(nf * (float)(k + 1)); mx = fmaxf(mx, l[k]); }
    float sum = 0.0f;
    for (int k = 0; k < q; ++k) { l[k] = __expf(l[k] - mx); sum += l[k]; }
    for (int k = 0; k < q; ++k) o[base + k] = l[k] / sum;
    base += q;
  }
  for (int c = 9; c < 16; ++c) o[c] = 0.0f;
}

__global__ void bp_edge_kernel(const int* __restrict__ src, const int* __restrict__ dst,
                               const float* __restrict__ w, const float* __restrict__ s,
                               float* __restrict__ field) {
  int e = blockIdx.x * blockDim.x + threadIdx.x;
  if (e >= EE) return;
  int si = src[e], di = dst[e];
  float we = w[e];
  const float4* s4 = (const float4*)(s + (size_t)si * 16);
  float4 v0 = s4[0], v1 = s4[1];
  float v8 = s[(size_t)si * 16 + 8];
  float* fr = field + (size_t)di * 16;
  atomicAdd(fr + 0, we * v0.x); atomicAdd(fr + 1, we * v0.y);
  atomicAdd(fr + 2, we * v0.z); atomicAdd(fr + 3, we * v0.w);
  atomicAdd(fr + 4, we * v1.x); atomicAdd(fr + 5, we * v1.y);
  atomicAdd(fr + 6, we * v1.z); atomicAdd(fr + 7, we * v1.w);
  atomicAdd(fr + 8, we * v8);
}

__global__ void bp_softmax_kernel(const float* __restrict__ field, float* __restrict__ s) {
  int n = blockIdx.x * blockDim.x + threadIdx.x;
  if (n >= NN) return;
  const float* f = field + (size_t)n * 16;
  float* o = s + (size_t)n * 16;
  const int offs[4] = {0, 2, 5, 9};
  for (int g = 0; g < 3; ++g) {
    int a = offs[g], b = offs[g + 1];
    float mx = -1e30f;
    for (int c = a; c < b; ++c) mx = fmaxf(mx, f[c]);
    float ex[4]; float sum = 0.0f;
    for (int c = a; c < b; ++c) { ex[c - a] = __expf(f[c] - mx); sum += ex[c - a]; }
    for (int c = a; c < b; ++c) o[c] = ex[c - a] / sum;
  }
  for (int c = 9; c < 16; ++c) o[c] = 0.0f;
}

// ---------------- modularity ----------------
// scal: [0]=m [1..3]=ein  [4..15]=aq(g*4+c)  [16..27]=bq  [28]=reg
__global__ void edge_stats_kernel(const int* __restrict__ src, const int* __restrict__ dst,
                                  const float* __restrict__ w, const float* __restrict__ s,
                                  float* __restrict__ dout, float* __restrict__ din,
                                  float* __restrict__ scal) {
  __shared__ float red[4][256];
  int t = threadIdx.x;
  int e = blockIdx.x * blockDim.x + t;
  float mw = 0, e2 = 0, e3 = 0, e4 = 0;
  if (e < EE) {
    int si = src[e], di = dst[e];
    float we = w[e];
    atomicAdd(dout + si, we);
    atomicAdd(din + di, we);
    const float* a = s + (size_t)si * 16;
    const float* b = s + (size_t)di * 16;
    mw = we;
    e2 = we * (a[0] * b[0] + a[1] * b[1]);
    e3 = we * (a[2] * b[2] + a[3] * b[3] + a[4] * b[4]);
    e4 = we * (a[5] * b[5] + a[6] * b[6] + a[7] * b[7] + a[8] * b[8]);
  }
  red[0][t] = mw; red[1][t] = e2; red[2][t] = e3; red[3][t] = e4;
  __syncthreads();
  for (int ofs = 128; ofs > 0; ofs >>= 1) {
    if (t < ofs) { red[0][t] += red[0][t + ofs]; red[1][t] += red[1][t + ofs];
                   red[2][t] += red[2][t + ofs]; red[3][t] += red[3][t + ofs]; }
    __syncthreads();
  }
  if (t == 0) {
    atomicAdd(scal + 0, red[0][0]); atomicAdd(scal + 1, red[1][0]);
    atomicAdd(scal + 2, red[2][0]); atomicAdd(scal + 3, red[3][0]);
  }
}

__global__ void aqbq_kernel(const float* __restrict__ s, const float* __restrict__ dout,
                            const float* __restrict__ din, float* __restrict__ scal) {
  __shared__ float acc[24];
  int t = threadIdx.x;
  if (t < 24) acc[t] = 0.0f;
  __syncthreads();
  int n = blockIdx.x * blockDim.x + t;
  if (n < NN) {
    float dv = dout[n], iv = din[n];
    const float* sr = s + (size_t)n * 16;
    const int offs[4] = {0, 2, 5, 9};
    for (int g = 0; g < 3; ++g)
      for (int c = offs[g]; c < offs[g + 1]; ++c) {
        atomicAdd(&acc[g * 4 + (c - offs[g])], dv * sr[c]);
        atomicAdd(&acc[12 + g * 4 + (c - offs[g])], iv * sr[c]);
      }
  }
  __syncthreads();
  if (t < 24) atomicAdd(scal + 4 + t, acc[t]);
}

__global__ void reg_kernel(float* __restrict__ scal) {
  if (threadIdx.x == 0 && blockIdx.x == 0) {
    float m = scal[0];
    float reg = 0.0f;
    for (int g = 0; g < 3; ++g) {
      float dd = 0.0f;
      for (int c = 0; c < 4; ++c) dd += scal[4 + g * 4 + c] * scal[16 + g * 4 + c];
      reg += scal[1 + g] / m - dd / (m * m);
    }
    scal[28] = reg;
  }
}

// ---------------- pooling MLP: s = softmax(relu(feat@W1+b1)@W2+b2), WMMA ----------------
__global__ void pf_kernel(const float* __restrict__ sbp, const float* __restrict__ W1,
                          const float* __restrict__ b1, const float* __restrict__ W2,
                          const float* __restrict__ b2, float* __restrict__ sc) {
  __shared__ _Float16 hstage[4][16 * 128];
  __shared__ float ostage[4][16 * 16];
  int wave = threadIdx.x >> 5, lane = threadIdx.x & 31;
  int tile = blockIdx.x * 4 + wave;
  int row = lane & 15, hi = lane >> 4, col = lane & 15;
  int node = tile * 16 + row;
  // A1: elements 0..7 load K = hi*8..hi*8+7 (always < 16); elements 8..15 are K>=16 -> zero
  v16h a1;
  {
    const float* sr = sbp + (size_t)node * 16 + hi * 8;
    #pragma unroll
    for (int p = 0; p < 4; ++p) {
      float2 v = *(const float2*)(sr + 2 * p);
      a1[2 * p + 0] = (_Float16)v.x;
      a1[2 * p + 1] = (_Float16)v.y;
    }
    #pragma unroll
    for (int e = 8; e < 16; ++e) a1[e] = (_Float16)0.0f;
  }
  _Float16* H = hstage[wave];
  #pragma unroll
  for (int t = 0; t < 7; ++t) {
    int c = t * 16 + col;
    int cc = (c < 100) ? c : 99;
    v16h b;
    #pragma unroll
    for (int e = 0; e < 16; ++e) {
      int k = hi * 8 + (e < 8 ? e : e + 8);
      int kc = (k < 9) ? k : 8;
      float w = W1[kc * 100 + cc];
      b[e] = (_Float16)((k < 9 && c < 100) ? w : 0.0f);
    }
    v8f z = {};
    v8f h = __builtin_amdgcn_wmma_f32_16x16x32_f16(false, a1, false, b, (short)0, z, false, false);
    float bias = b1[cc];
    #pragma unroll
    for (int r = 0; r < 8; ++r) {
      float v = fmaxf(h[r] + bias, 0.0f);
      H[(r + 8 * hi) * 128 + c] = (_Float16)((c < 100) ? v : 0.0f);
    }
  }
  #pragma unroll
  for (int r = 0; r < 8; ++r) H[(r + 8 * hi) * 128 + 112 + col] = (_Float16)0.0f;
  __syncthreads();
  v8f acc = {};
  int cc2 = (col < 10) ? col : 9;
  #pragma unroll
  for (int ks = 0; ks < 4; ++ks) {
    v16h a2;
    #pragma unroll
    for (int e = 0; e < 16; ++e) {
      int k = ks * 32 + hi * 8 + (e < 8 ? e : e + 8);
      a2[e] = H[row * 128 + k];
    }
    v16h b;
    #pragma unroll
    for (int e = 0; e < 16; ++e) {
      int k = ks * 32 + hi * 8 + (e < 8 ? e : e + 8);
      int kc = (k < 100) ? k : 99;
      float w = W2[kc * 10 + cc2];
      b[e] = (_Float16)((k < 100 && col < 10) ? w : 0.0f);
    }
    acc = __builtin_amdgcn_wmma_f32_16x16x32_f16(false, a2, false, b, (short)0, acc, false, false);
  }
  float bias2 = (col < 10) ? b2[cc2] : 0.0f;
  float* O = ostage[wave];
  #pragma unroll
  for (int r = 0; r < 8; ++r) O[(r + 8 * hi) * 16 + col] = acc[r] + bias2;
  __syncthreads();
  if (lane < 16) {
    float* orow = O + lane * 16;
    float mx = -1e30f;
    for (int c = 0; c < 10; ++c) mx = fmaxf(mx, orow[c]);
    float ex[10]; float sum = 0.0f;
    for (int c = 0; c < 10; ++c) { ex[c] = __expf(orow[c] - mx); sum += ex[c]; }
    float* dstp = sc + (size_t)(tile * 16 + lane) * 16;
    for (int c = 0; c < 10; ++c) dstp[c] = ex[c] / sum;
    for (int c = 10; c < 16; ++c) dstp[c] = 0.0f;
  }
}

// ---------------- p1_x = s^T x1 per graph (WMMA, 8 waves per graph, atomic accumulate) ----------------
__global__ void pool_px_kernel(const float* __restrict__ sc, const float* __restrict__ x1c,
                               float* __restrict__ p1x) {
  int b = blockIdx.x;           // graph
  int chunk = blockIdx.y;       // 0..7 -> 128 nodes each
  int lane = threadIdx.x & 31;
  int row = lane & 15, hi = lane >> 4, col = lane & 15;
  v8f zz = {};
  v8f acc[6];
  #pragma unroll
  for (int t = 0; t < 6; ++t) acc[t] = zz;
  int n0 = chunk * 128;
  for (int nb = n0; nb < n0 + 128; nb += 32) {
    v16h a;  // A[m=cluster, k=node] = s[node, cluster] (clusters>=10 are zero)
    #pragma unroll
    for (int e = 0; e < 16; ++e) {
      int n = nb + hi * 8 + (e < 8 ? e : e + 8);
      a[e] = (_Float16)sc[((size_t)b * NPGC + n) * 16 + row];
    }
    #pragma unroll
    for (int t = 0; t < 6; ++t) {
      v16h bf;
      #pragma unroll
      for (int e = 0; e < 16; ++e) {
        int n = nb + hi * 8 + (e < 8 ? e : e + 8);
        bf[e] = (_Float16)x1c[((size_t)b * NPGC + n) * 96 + t * 16 + col];  // cols 90..95 zero
      }
      acc[t] = __builtin_amdgcn_wmma_f32_16x16x32_f16(false, a, false, bf, (short)0, acc[t], false, false);
    }
  }
  #pragma unroll
  for (int t = 0; t < 6; ++t)
    #pragma unroll
    for (int r = 0; r < 8; ++r)
      atomicAdd(&p1x[((size_t)b * 16 + r + 8 * hi) * 96 + t * 16 + col], acc[t][r]);
}

// ---------------- p1_adj = sum_e w * s_src (x) s_dst ----------------
// Per-wave LDS accumulators + lane-rotated (c,k) order to break same-address serialization.
__global__ void padj_kernel(const int* __restrict__ src, const int* __restrict__ dst,
                            const float* __restrict__ w, const float* __restrict__ sc,
                            float* __restrict__ padj) {
  __shared__ float acc[8][100];
  int b = blockIdx.x, t = threadIdx.x;  // 256 threads = 8 waves
  int wave = t >> 5, lane = t & 31;
  int lc = lane % 10;
  for (int i = t; i < 800; i += 256) acc[i / 100][i % 100] = 0.0f;
  __syncthreads();
  float* wacc = acc[wave];
  for (int e = t; e < EPG; e += 256) {
    int ge = b * EPG + e;
    int si = src[ge], di = dst[ge];
    float we = w[ge];
    const float* a = sc + (size_t)si * 16;
    const float* d = sc + (size_t)di * 16;
    // load rotated: ssr[ci] = s_src[(ci+lc)%10], sdr[ki] = s_dst[(ki+lc)%10]
    float ssr[10], sdr[10];
    #pragma unroll
    for (int ci = 0; ci < 10; ++ci) {
      int c = ci + lc; if (c >= 10) c -= 10;
      ssr[ci] = a[c]; sdr[ci] = d[c];
    }
    #pragma unroll
    for (int ci = 0; ci < 10; ++ci) {
      int c = ci + lc; if (c >= 10) c -= 10;
      float wc = we * ssr[ci];
      #pragma unroll
      for (int ki = 0; ki < 10; ++ki) {
        int k = ki + lc; if (k >= 10) k -= 10;
        atomicAdd(&wacc[c * 10 + k], wc * sdr[ki]);
      }
    }
  }
  __syncthreads();
  for (int i = t; i < 100; i += 256) {
    float sum = 0.0f;
    #pragma unroll
    for (int wv = 0; wv < 8; ++wv) sum += acc[wv][i];
    padj[(size_t)b * 100 + i] = sum;
  }
}

// ---------------- stage-2 dense GIN + max over clusters ----------------
__global__ void stage2_kernel(const float* __restrict__ p1x, const float* __restrict__ padj,
                              const float* __restrict__ W11, const float* __restrict__ W12,
                              const float* __restrict__ W21, const float* __restrict__ W22,
                              const float* __restrict__ W31, const float* __restrict__ W32,
                              float* __restrict__ x2out) {
  __shared__ float msk[10][10];
  __shared__ float hcur[10][96];
  __shared__ float hs[10][96];
  __shared__ float tmp[10][32];
  __shared__ float xcat[10][96];
  int b = blockIdx.x, t = threadIdx.x;
  for (int i = t; i < 100; i += 256) {
    float v = padj[(size_t)b * 100 + i];
    msk[i / 10][i % 10] = (v != 0.0f) ? 1.0f : 0.0f;
  }
  for (int i = t; i < 960; i += 256) {
    int r = i / 96, c = i % 96;
    hcur[r][c] = (c < 90) ? p1x[((size_t)b * 16 + r) * 96 + c] : 0.0f;
  }
  __syncthreads();
  const float* W1s[3] = {W11, W21, W31};
  const float* W2s[3] = {W12, W22, W32};
  for (int L = 0; L < 3; ++L) {
    int Fin = (L == 0) ? 90 : 30;
    for (int i = t; i < 10 * Fin; i += 256) {
      int r = i / Fin, c = i % Fin;
      float sum = hcur[r][c];
      for (int j = 0; j < 10; ++j) sum += msk[r][j] * hcur[j][c];
      hs[r][c] = sum;
    }
    __syncthreads();
    for (int i = t; i < 300; i += 256) {
      int r = i / 30, h = i % 30;
      float sum = 0.0f;
      const float* W1 = W1s[L];
      for (int f = 0; f < Fin; ++f) sum += hs[r][f] * W1[f * 30 + h];
      tmp[r][h] = fmaxf(sum, 0.0f);
    }
    __syncthreads();
    for (int i = t; i < 300; i += 256) {
      int r = i / 30, o = i % 30;
      float sum = 0.0f;
      const float* W2 = W2s[L];
      for (int h = 0; h < 30; ++h) sum += tmp[r][h] * W2[h * 30 + o];
      xcat[r][L * 30 + o] = sum;
    }
    __syncthreads();
    for (int i = t; i < 300; i += 256) {
      int r = i / 30, o = i % 30;
      hcur[r][o] = xcat[r][L * 30 + o];
    }
    __syncthreads();
  }
  for (int f = t; f < 90; f += 256) {
    float mx = -1e30f;
    for (int r = 0; r < 10; ++r) mx = fmaxf(mx, xcat[r][f]);
    x2out[(size_t)b * 96 + f] = mx;
  }
}

// ---------------- x1_out = max over nodes per graph (two-stage) ----------------
__global__ void x1out_part_kernel(const float* __restrict__ x1c, float* __restrict__ part) {
  int b = blockIdx.x, ch = blockIdx.y, f = threadIdx.x;  // blockDim 96
  float mx = -1e30f;
  int n0 = ch * 128;
  for (int n = n0; n < n0 + 128; ++n)
    mx = fmaxf(mx, x1c[((size_t)b * NPGC + n) * 96 + f]);
  part[((size_t)b * 8 + ch) * 96 + f] = mx;
}

__global__ void x1out_reduce_kernel(const float* __restrict__ part, float* __restrict__ x1out) {
  int b = blockIdx.x, f = threadIdx.x;  // blockDim 96
  float mx = -1e30f;
  #pragma unroll
  for (int ch = 0; ch < 8; ++ch)
    mx = fmaxf(mx, part[((size_t)b * 8 + ch) * 96 + f]);
  x1out[(size_t)b * 96 + f] = mx;
}

// ---------------- BN + MLP head (single block) ----------------
__global__ void head_kernel(const float* __restrict__ x1out, const float* __restrict__ x2out,
                            const float* __restrict__ g1, const float* __restrict__ bt1,
                            const float* __restrict__ fW1, const float* __restrict__ fb1,
                            const float* __restrict__ g2, const float* __restrict__ bt2,
                            const float* __restrict__ fW2, const float* __restrict__ fb2,
                            const float* __restrict__ scal, float* __restrict__ out) {
  __shared__ float h[64][180];
  __shared__ float h2[64][50];
  int t = threadIdx.x;  // 256
  for (int i = t; i < 64 * 180; i += 256) {
    int r = i / 180, c = i % 180;
    h[r][c] = (c < 90) ? x1out[(size_t)r * 96 + c] : x2out[(size_t)r * 96 + (c - 90)];
  }
  __syncthreads();
  for (int c = t; c < 180; c += 256) {
    float mu = 0.0f;
    for (int r = 0; r < 64; ++r) mu += h[r][c];
    mu *= (1.0f / 64.0f);
    float v = 0.0f;
    for (int r = 0; r < 64; ++r) { float d = h[r][c] - mu; v += d * d; }
    v *= (1.0f / 64.0f);
    float inv = rsqrtf(v + 1e-5f);
    for (int r = 0; r < 64; ++r) {
      float y = g1[c] * (h[r][c] - mu) * inv + bt1[c];
      h[r][c] = fmaxf(y, 0.0f);
    }
  }
  __syncthreads();
  for (int i = t; i < 64 * 50; i += 256) {
    int r = i / 50, k = i % 50;
    float sum = fb1[k];
    for (int j = 0; j < 180; ++j) sum += h[r][j] * fW1[j * 50 + k];
    h2[r][k] = sum;
  }
  __syncthreads();
  for (int k = t; k < 50; k += 256) {
    float mu = 0.0f;
    for (int r = 0; r < 64; ++r) mu += h2[r][k];
    mu *= (1.0f / 64.0f);
    float v = 0.0f;
    for (int r = 0; r < 64; ++r) { float d = h2[r][k] - mu; v += d * d; }
    v *= (1.0f / 64.0f);
    float inv = rsqrtf(v + 1e-5f);
    for (int r = 0; r < 64; ++r) {
      float y = g2[k] * (h2[r][k] - mu) * inv + bt2[k];
      h2[r][k] = fmaxf(y, 0.0f);
    }
  }
  __syncthreads();
  for (int i = t; i < 128; i += 256) {
    int r = i / 2, o = i % 2;
    float sum = fb2[o];
    for (int k = 0; k < 50; ++k) sum += h2[r][k] * fW2[k * 2 + o];
    out[i] = sum;
  }
  if (t == 0) out[128] = scal[28];
}

// ---------------- host launch ----------------
extern "C" void kernel_launch(void* const* d_in, const int* in_sizes, int n_in,
                              void* d_out, int out_size, void* d_ws, size_t ws_size,
                              hipStream_t stream) {
  const float* x     = (const float*)d_in[0];
  const int*   ei    = (const int*)d_in[1];
  const float* ea    = (const float*)d_in[2];
  const float* c11W1 = (const float*)d_in[3];
  const float* c11W2 = (const float*)d_in[4];
  const float* c12W1 = (const float*)d_in[5];
  const float* c12W2 = (const float*)d_in[6];
  const float* c13W1 = (const float*)d_in[7];
  const float* c13W2 = (const float*)d_in[8];
  const float* pfW1  = (const float*)d_in[9];
  const float* pfb1  = (const float*)d_in[10];
  const float* pfW2  = (const float*)d_in[11];
  const float* pfb2  = (const float*)d_in[12];
  const float* c21W1 = (const float*)d_in[13];
  const float* c21W2 = (const float*)d_in[14];
  const float* c22W1 = (const float*)d_in[15];
  const float* c22W2 = (const float*)d_in[16];
  const float* c23W1 = (const float*)d_in[17];
  const float* c23W2 = (const float*)d_in[18];
  const float* bn1g  = (const float*)d_in[19];
  const float* bn1b  = (const float*)d_in[20];
  const float* fW1   = (const float*)d_in[21];
  const float* fb1   = (const float*)d_in[22];
  const float* bn2g  = (const float*)d_in[23];
  const float* bn2b  = (const float*)d_in[24];
  const float* fW2   = (const float*)d_in[25];
  const float* fb2   = (const float*)d_in[26];

  const int* src = ei;
  const int* dst = ei + EE;

  float* ws   = (float*)d_ws;
  float* agg0 = ws + OFF_AGG0;
  float* x1c  = ws + OFF_X1C;
  float* aggH = ws + OFF_AGGH;
  float* sbp  = ws + OFF_SBP;
  float* fld  = ws + OFF_FLD;
  float* sc   = ws + OFF_SC;
  float* dou  = ws + OFF_DOUT;
  float* din  = ws + OFF_DIN;
  float* p1x  = ws + OFF_P1X;
  float* padj = ws + OFF_PADJ;
  float* x2o  = ws + OFF_X2O;
  float* x1o  = ws + OFF_X1O;
  float* scal = ws + OFF_SCAL;
  float* x1p  = ws + OFF_X1P;
  float* out  = (float*)d_out;

  auto Z = [&](float* p, size_t n) {
    zero_kernel<<<(unsigned)((n + 255) / 256), 256, 0, stream>>>(p, n);
  };

  const int edgeBlocks = EE / 256;     // 8192
  const int nodeBlocks = NN / 256;     // 256
  const int tileBlocks = NN / 64;      // 1024 (4 waves/block, 16 nodes/wave)

  // ---- stage-1 GIN (concat output in x1c, stride 96) ----
  Z(agg0, (size_t)NN * 64);
  scatter_add_kernel<64, 64><<<edgeBlocks, 256, 0, stream>>>(src, dst, x, agg0);
  gin_mlp_kernel<64, 64, 64, 0><<<tileBlocks, 128, 0, stream>>>(x, agg0, c11W1, c11W2, x1c);

  Z(aggH, (size_t)NN * 32);
  scatter_add_kernel<96, 32><<<edgeBlocks, 256, 0, stream>>>(src, dst, x1c, aggH);
  gin_mlp_kernel<96, 32, 30, 30><<<tileBlocks, 128, 0, stream>>>(x1c, aggH, c12W1, c12W2, x1c);

  Z(aggH, (size_t)NN * 32);
  scatter_add_kernel<96, 32><<<edgeBlocks, 256, 0, stream>>>(src, dst, x1c + 30, aggH);
  gin_mlp_kernel<96, 32, 30, 60><<<tileBlocks, 128, 0, stream>>>(x1c + 30, aggH, c13W1, c13W2, x1c);

  // ---- belief propagation (q = 2,3,4 fused in one [N,16] buffer) ----
  bp_init_kernel<<<nodeBlocks, 256, 0, stream>>>(sbp);
  for (int it = 0; it < 5; ++it) {
    Z(fld, (size_t)NN * 16);
    bp_edge_kernel<<<edgeBlocks, 256, 0, stream>>>(src, dst, ea, sbp, fld);
    bp_softmax_kernel<<<nodeBlocks, 256, 0, stream>>>(fld, sbp);
  }

  // ---- modularity regularizer ----
  Z(dou, (size_t)2 * NN);   // dout + din contiguous
  Z(scal, 32);
  edge_stats_kernel<<<edgeBlocks, 256, 0, stream>>>(src, dst, ea, sbp, dou, din, scal);
  aqbq_kernel<<<nodeBlocks, 256, 0, stream>>>(sbp, dou, din, scal);
  reg_kernel<<<1, 32, 0, stream>>>(scal);

  // ---- cluster assignment MLP ----
  pf_kernel<<<tileBlocks, 128, 0, stream>>>(sbp, pfW1, pfb1, pfW2, pfb2, sc);

  // ---- DiffPool ----
  Z(p1x, (size_t)64 * 16 * 96);
  pool_px_kernel<<<dim3(BB, 8), 32, 0, stream>>>(sc, x1c, p1x);
  padj_kernel<<<BB, 256, 0, stream>>>(src, dst, ea, sc, padj);

  // ---- stage-2 GIN + pools + head ----
  stage2_kernel<<<BB, 256, 0, stream>>>(p1x, padj, c21W1, c21W2, c22W1, c22W2,
                                        c23W1, c23W2, x2o);
  x1out_part_kernel<<<dim3(BB, 8), 96, 0, stream>>>(x1c, x1p);
  x1out_reduce_kernel<<<BB, 96, 0, stream>>>(x1p, x1o);
  head_kernel<<<1, 256, 0, stream>>>(x1o, x2o, bn1g, bn1b, fW1, fb1, bn2g, bn2b,
                                     fW2, fb2, scal, out);
}
